// RPNModule_42606075576322
// MI455X (gfx1250) — matrix operand order repaired and checked by reading, not compiled, not run on hardware
//
#include <hip/hip_runtime.h>
#include <hip/hip_bf16.h>
#include <stdint.h>

#define NIMG 2
#define CIN 256
#define HFD 64
#define WFD 64
#define AANCH 24
#define PRE_NMS 1000
#define POST_NMS 500
#define NMS_TH 0.7f
#define PIX_PER_IMG (HFD * WFD)      /* 4096  */
#define NPIX (NIMG * PIX_PER_IMG)    /* 8192  */
#define NPROP (PIX_PER_IMG * AANCH)  /* 98304 */
#define SORTN 131072
#define DEGC 0.017453292519943295f
#define CLIPC 4.1351665567423560f    /* log(1000/16) */

typedef __attribute__((ext_vector_type(16))) __bf16 v16bf;
typedef __attribute__((ext_vector_type(8)))  float  v8f;

union Frag { v16bf v; uint4 q[2]; unsigned short s[16]; };

__device__ __forceinline__ unsigned short f2bf(float f) {
  unsigned u = __float_as_uint(f);
  u += 0x7FFFu + ((u >> 16) & 1u);          // round-to-nearest-even
  return (unsigned short)(u >> 16);
}

// ---------------------------------------------------------------------------
// Prep 1: pack conv weights into WMMA B-fragment lane order, bf16.
// B tile (kt = tap*8 + kc/32, ntile = out-col/16): per lane 16 elems:
//   N = ntile*16 + (lane&15), K = ((lane<16)?0:16)+e  -> contiguous 32B/lane.
// ---------------------------------------------------------------------------
__global__ void pack_conv_w(const float* __restrict__ w,
                            unsigned short* __restrict__ wpack) {
  int gid = blockIdx.x * blockDim.x + threadIdx.x;   // 72*16*32*16 = 589824
  if (gid >= 72 * 16 * 512) return;
  int e    = gid & 15;
  int lane = (gid >> 4) & 31;
  int tile = gid >> 9;
  int ntile = tile & 15;          // 0..15 (16 out-cols each -> 256 outs)
  int kt    = tile >> 4;          // 0..71
  int tap = kt >> 3;
  int kc  = (kt & 7) * 32;
  int N  = ntile * 16 + (lane & 15);
  int ic = kc + ((lane < 16) ? 0 : 16) + e;
  wpack[gid] = f2bf(w[((size_t)N * CIN + ic) * 9 + tap]);
}

// Prep 2: pack fused head weights (cls 24 rows + bbox 120 rows = 144 outs).
__global__ void pack_head_w(const float* __restrict__ cls_w,
                            const float* __restrict__ bbox_w,
                            unsigned short* __restrict__ hpack) {
  int gid = blockIdx.x * blockDim.x + threadIdx.x;   // 8*9*32*16 = 36864
  if (gid >= 8 * 9 * 512) return;
  int e    = gid & 15;
  int lane = (gid >> 4) & 31;
  int tile = gid >> 9;
  int ntile = tile % 9;
  int kt    = tile / 9;
  int out = ntile * 16 + (lane & 15);
  int ic  = kt * 32 + ((lane < 16) ? 0 : 16) + e;
  float v = (out < AANCH) ? cls_w[(size_t)out * CIN + ic]
                          : bbox_w[(size_t)(out - AANCH) * CIN + ic];
  hpack[gid] = f2bf(v);
}

// ---------------------------------------------------------------------------
// Kernel 1: 3x3 conv (SAME) + bias + ReLU as implicit-GEMM WMMA.
// Tile: one image row (64 px) x 64 out-ch per 256-thread WG (8 waves).
// A staged in LDS (zero-padded borders), B read packed from L2.
// ---------------------------------------------------------------------------
__global__ __launch_bounds__(256) void conv3x3_wmma(
    const float* __restrict__ feat, const unsigned short* __restrict__ wpack,
    const float* __restrict__ bias, unsigned short* __restrict__ tout) {
  __shared__ __align__(16) unsigned short As[64][32];  // [w-pixel][in-ch chunk]
  const int octile = blockIdx.x;          // 0..3   (64 out-ch each)
  const int bm     = blockIdx.y;          // 0..127 (img*64 + h)
  const int n = bm >> 6;
  const int h = bm & 63;
  const int tid  = threadIdx.x;
  const int wave = tid >> 5;
  const int lane = tid & 31;
  const int mt  = wave & 3;               // row sub-tile 0..3
  const int nt0 = (wave >> 2) * 2;        // col sub-tiles {nt0, nt0+1}
  const int M   = lane & 15;
  const int kb  = (lane < 16) ? 0 : 16;   // A K-base (contiguous, matches packed B)

  v8f acc0 = {0.f,0.f,0.f,0.f,0.f,0.f,0.f,0.f};
  v8f acc1 = {0.f,0.f,0.f,0.f,0.f,0.f,0.f,0.f};

  const int ntg0 = octile * 4 + nt0;      // global 16-col tile ids
  const int ntg1 = ntg0 + 1;

  for (int tap = 0; tap < 9; ++tap) {
    const int ky = tap / 3, kx = tap % 3;
    const int y = h + ky - 1;
    for (int kcc = 0; kcc < 8; ++kcc) {   // 32-channel chunks
      const int kc = kcc * 32;
      // stage A: 64 pixels x 32 channels, pixel-fastest for coalescing
      for (int it = 0; it < 8; ++it) {
        int e = it * 256 + tid;
        int row = e & 63, col = e >> 6;
        int x = row + kx - 1;
        float v = 0.0f;
        if (y >= 0 && y < HFD && x >= 0 && x < WFD)
          v = feat[(((n * CIN + (kc + col)) * HFD + y) * WFD) + x];
        As[row][col] = f2bf(v);
      }
      __syncthreads();

      // A fragment: two 16B LDS loads
      Frag a;
      const unsigned short* arow = &As[mt * 16 + M][0];
      a.q[0] = *(const uint4*)(arow + kb);
      a.q[1] = *(const uint4*)(arow + kb + 8);

      // B fragments: two 16B global loads each, from packed L2-resident weights
      const int kt = tap * 8 + kcc;
      Frag b0, b1;
      const uint4* bp0 = (const uint4*)(wpack + (((size_t)kt * 16 + ntg0) * 32 + lane) * 16);
      const uint4* bp1 = (const uint4*)(wpack + (((size_t)kt * 16 + ntg1) * 32 + lane) * 16);
      b0.q[0] = bp0[0]; b0.q[1] = bp0[1];
      b1.q[0] = bp1[0]; b1.q[1] = bp1[1];
      // prefetch next K-tile's B operands
      if (kt < 71) {
        __builtin_prefetch(wpack + (((size_t)(kt + 1) * 16 + ntg0) * 32 + lane) * 16, 0, 3);
        __builtin_prefetch(wpack + (((size_t)(kt + 1) * 16 + ntg1) * 32 + lane) * 16, 0, 3);
      }

      acc0 = __builtin_amdgcn_wmma_f32_16x16x32_bf16(false, a.v, false, b0.v,
                                                     (short)0, acc0, false, false);
      acc1 = __builtin_amdgcn_wmma_f32_16x16x32_bf16(false, a.v, false, b1.v,
                                                     (short)0, acc1, false, false);
      __syncthreads();
    }
  }
  // epilogue: bias + ReLU, store bf16 activations pixel-major [pix][256]
  const int M0 = (lane < 16) ? 0 : 8;
  const int oc0 = octile * 64 + nt0 * 16 + M;
  const int oc1 = oc0 + 16;
  const float bi0 = bias[oc0], bi1 = bias[oc1];
  #pragma unroll
  for (int r = 0; r < 8; ++r) {
    int wp  = mt * 16 + r + M0;
    int row = n * PIX_PER_IMG + h * WFD + wp;
    float v0 = acc0[r] + bi0; v0 = v0 > 0.f ? v0 : 0.f;
    float v1 = acc1[r] + bi1; v1 = v1 > 0.f ? v1 : 0.f;
    tout[(size_t)row * CIN + oc0] = f2bf(v0);
    tout[(size_t)row * CIN + oc1] = f2bf(v1);
  }
}

// ---------------------------------------------------------------------------
// Kernel 2: fused 1x1 heads (cls 24 + bbox 120 = 144 outs) as WMMA GEMM.
// 1 wave per WG, 16 pixels x 16 outs tile; A and B both straight from global
// as 16B vector loads (activations pixel-major bf16, weights pre-packed).
// ---------------------------------------------------------------------------
__global__ __launch_bounds__(32) void head_wmma(
    const unsigned short* __restrict__ tin,
    const unsigned short* __restrict__ hpack,
    const float* __restrict__ cls_b, const float* __restrict__ bbox_b,
    float* __restrict__ breg, unsigned long long* __restrict__ keys) {
  const int mt = blockIdx.x;    // 0..511 pixel tile
  const int nt = blockIdx.y;    // 0..8   out tile
  const int lane = threadIdx.x;
  const int N = lane & 15;
  const int pix = mt * 16 + N;
  const int out = nt * 16 + N;
  const unsigned short* arow = tin + (size_t)pix * CIN;
  const int kb = (lane < 16) ? 0 : 16;
  v8f acc = {0.f,0.f,0.f,0.f,0.f,0.f,0.f,0.f};
  #pragma unroll
  for (int kt = 0; kt < 8; ++kt) {
    Frag a, b;
    const unsigned short* ap = arow + kt * 32 + kb;
    a.q[0] = *(const uint4*)(ap);
    a.q[1] = *(const uint4*)(ap + 8);
    const uint4* bp = (const uint4*)(hpack + (((size_t)kt * 9 + nt) * 32 + lane) * 16);
    b.q[0] = bp[0]; b.q[1] = bp[1];
    acc = __builtin_amdgcn_wmma_f32_16x16x32_bf16(false, a.v, false, b.v,
                                                  (short)0, acc, false, false);
  }
  const int M0 = (lane < 16) ? 0 : 8;
  #pragma unroll
  for (int r = 0; r < 8; ++r) {
    int p  = mt * 16 + r + M0;
    int n  = p >> 12;
    int lp = p & 4095;
    float v = acc[r];
    if (out < AANCH) {
      float sc = 1.0f / (1.0f + expf(-(v + cls_b[out])));
      unsigned idx = (unsigned)(lp * AANCH + out);
      unsigned long long key =
          ((unsigned long long)__float_as_uint(sc) << 32) |
          (unsigned long long)(0xFFFFFFFFu - idx);
      keys[(size_t)n * SORTN + idx] = key;
    } else {
      int j = out - AANCH;
      int a = j / 5, c = j % 5;
      breg[((size_t)n * NPROP + lp * AANCH + a) * 5 + c] = v + bbox_b[j];
    }
  }
}

// ---------------------------------------------------------------------------
// Sort / decode / IoU / NMS
// ---------------------------------------------------------------------------
__global__ void zero_keys(unsigned long long* keys) {
  int gid = blockIdx.x * blockDim.x + threadIdx.x;
  if (gid < NIMG * SORTN) keys[gid] = 0ull;
}

__global__ void bitonic_kernel(unsigned long long* keys, int k, int j) {
  unsigned gid = blockIdx.x * blockDim.x + threadIdx.x;  // 0..262143
  unsigned n = gid >> 17;
  unsigned i = gid & (SORTN - 1);
  unsigned ixj = i ^ (unsigned)j;
  if (ixj > i) {
    unsigned long long* base = keys + (size_t)n * SORTN;
    unsigned long long a = base[i], b = base[ixj];
    bool desc = ((i & (unsigned)k) == 0);
    if (desc ? (a < b) : (a > b)) { base[i] = b; base[ixj] = a; }
  }
}

__global__ void decode_kernel(const unsigned long long* __restrict__ keys,
                              const float* __restrict__ breg,
                              const float* __restrict__ anchors,
                              float* __restrict__ out) {
  int tid = blockIdx.x * blockDim.x + threadIdx.x;
  if (tid >= NIMG * PRE_NMS) return;
  int n = tid / PRE_NMS, i = tid % PRE_NMS;
  unsigned long long key = keys[(size_t)n * SORTN + i];
  float score = __uint_as_float((unsigned)(key >> 32));
  unsigned idx = 0xFFFFFFFFu - (unsigned)(key & 0xFFFFFFFFull);
  const float* anc = anchors + ((size_t)n * NPROP + idx) * 5;
  const float* cd  = breg    + ((size_t)n * NPROP + idx) * 5;
  float aw = anc[2], ah = anc[3];
  float* p = out + ((size_t)n * PRE_NMS + i) * 5;
  p[0] = cd[0] * aw + anc[0];
  p[1] = cd[1] * ah + anc[1];
  p[2] = expf(fminf(cd[2], CLIPC)) * aw;
  p[3] = expf(fminf(cd[3], CLIPC)) * ah;
  p[4] = anc[4] + cd[4] / DEGC;
  out[NIMG * PRE_NMS * 5 + n * PRE_NMS + i] = score;
}

__device__ void get_corners(const float* b, float* cx, float* cy) {
  float rad = b[4] * DEGC, c = cosf(rad), s = sinf(rad);
  const float dxs[4] = {0.5f, -0.5f, -0.5f, 0.5f};
  const float dys[4] = {0.5f, 0.5f, -0.5f, -0.5f};
  #pragma unroll
  for (int k = 0; k < 4; ++k) {
    float dx = dxs[k] * b[2], dy = dys[k] * b[3];
    cx[k] = b[0] + dx * c - dy * s;
    cy[k] = b[1] + dx * s + dy * c;
  }
}

__device__ void inside4(const float* px, const float* py, const float* b, bool* m) {
  float rad = b[4] * DEGC, c = cosf(rad), s = sinf(rad);
  #pragma unroll
  for (int k = 0; k < 4; ++k) {
    float rx = px[k] - b[0], ry = py[k] - b[1];
    float lx =  rx * c + ry * s;
    float ly = -rx * s + ry * c;
    m[k] = (fabsf(lx) <= b[2] * 0.5f + 1e-4f) && (fabsf(ly) <= b[3] * 0.5f + 1e-4f);
  }
}

__device__ float pair_iou(const float* A, const float* B) {
  float cax[4], cay[4], cbx[4], cby[4];
  get_corners(A, cax, cay);
  get_corners(B, cbx, cby);
  float rx[4], ry[4], sx[4], sy[4];
  #pragma unroll
  for (int i = 0; i < 4; ++i) {
    rx[i] = cax[(i + 1) & 3] - cax[i];  ry[i] = cay[(i + 1) & 3] - cay[i];
    sx[i] = cbx[(i + 1) & 3] - cbx[i];  sy[i] = cby[(i + 1) & 3] - cby[i];
  }
  float ptx[24], pty[24];
  bool  m[24];
  int p = 0;
  for (int i = 0; i < 4; ++i)
    for (int j = 0; j < 4; ++j) {
      float den  = rx[i] * sy[j] - ry[i] * sx[j];
      float safe = (fabsf(den) > 1e-8f) ? den : 1.0f;
      float qpx = cbx[j] - cax[i], qpy = cby[j] - cay[i];
      float t = (qpx * sy[j] - qpy * sx[j]) / safe;
      float u = (qpx * ry[i] - qpy * rx[i]) / safe;
      bool ok = (fabsf(den) > 1e-8f) && (t >= 0.f) && (t <= 1.f) &&
                (u >= 0.f) && (u <= 1.f);
      m[p]   = ok;
      ptx[p] = ok ? cax[i] + t * rx[i] : 0.f;
      pty[p] = ok ? cay[i] + t * ry[i] : 0.f;
      ++p;
    }
  bool ma[4], mb[4];
  inside4(cax, cay, B, ma);
  inside4(cbx, cby, A, mb);
  #pragma unroll
  for (int k = 0; k < 4; ++k) {
    m[16 + k] = ma[k]; ptx[16 + k] = cax[k]; pty[16 + k] = cay[k];
    m[20 + k] = mb[k]; ptx[20 + k] = cbx[k]; pty[20 + k] = cby[k];
  }
  int nv = 0; float s0x = 0.f, s0y = 0.f;
  for (int k = 0; k < 24; ++k)
    if (m[k]) { ++nv; s0x += ptx[k]; s0y += pty[k]; }
  int nvc = nv > 0 ? nv : 1;
  float c0x = s0x / (float)nvc, c0y = s0y / (float)nvc;
  float ang[24];
  for (int k = 0; k < 24; ++k)
    ang[k] = m[k] ? atan2f(pty[k] - c0y, ptx[k] - c0x) : 1e6f;
  int ord[24];
  for (int k = 0; k < 24; ++k) ord[k] = k;
  for (int k = 1; k < 24; ++k) {         // stable insertion sort (asc angle)
    int key = ord[k]; float ak = ang[key];
    int l = k - 1;
    while (l >= 0 && ang[ord[l]] > ak) { ord[l + 1] = ord[l]; --l; }
    ord[l + 1] = key;
  }
  float qx[24], qy[24];
  float fx = ptx[ord[0]], fy = pty[ord[0]];
  for (int k = 0; k < 24; ++k) {
    int o = ord[k];
    qx[k] = m[o] ? ptx[o] : fx;
    qy[k] = m[o] ? pty[o] : fy;
  }
  float area2 = 0.f;
  for (int k = 0; k < 24; ++k) {
    int k2 = (k + 1) % 24;
    area2 += qx[k] * qy[k2] - qy[k] * qx[k2];
  }
  float inter = 0.5f * fabsf(area2);
  float uni = A[2] * A[3] + B[2] * B[3] - inter;
  return inter / fmaxf(uni, 1e-6f);
}

__global__ void iou_kernel(const float* __restrict__ props, float* __restrict__ iou) {
  long gid = (long)blockIdx.x * blockDim.x + threadIdx.x;
  if (gid >= (long)NIMG * PRE_NMS * PRE_NMS) return;
  int n   = (int)(gid / ((long)PRE_NMS * PRE_NMS));
  int rem = (int)(gid % ((long)PRE_NMS * PRE_NMS));
  int i = rem / PRE_NMS, j = rem % PRE_NMS;
  const float* A = props + ((size_t)n * PRE_NMS + i) * 5;
  const float* B = props + ((size_t)n * PRE_NMS + j) * 5;
  iou[gid] = pair_iou(A, B);
}

__global__ __launch_bounds__(1024) void nms_kernel(const float* __restrict__ iou,
                                                   float* __restrict__ keep_out) {
  __shared__ int kept[PRE_NMS];
  int n = blockIdx.x, tid = threadIdx.x;
  for (int i = tid; i < PRE_NMS; i += 1024) kept[i] = 1;
  __syncthreads();
  const float* M = iou + (size_t)n * PRE_NMS * PRE_NMS;
  for (int i = 0; i < PRE_NMS; ++i) {
    if (kept[i]) {
      if (tid < PRE_NMS && tid > i && M[(size_t)i * PRE_NMS + tid] > NMS_TH)
        kept[tid] = 0;
    }
    __syncthreads();
  }
  if (tid == 0) {
    int r = 0;
    for (int i = 0; i < PRE_NMS; ++i)
      if (kept[i]) { ++r; if (r > POST_NMS) kept[i] = 0; }
  }
  __syncthreads();
  if (tid < PRE_NMS) keep_out[n * PRE_NMS + tid] = kept[tid] ? 1.0f : 0.0f;
}

// ---------------------------------------------------------------------------
extern "C" void kernel_launch(void* const* d_in, const int* in_sizes, int n_in,
                              void* d_out, int out_size, void* d_ws, size_t ws_size,
                              hipStream_t stream) {
  (void)in_sizes; (void)n_in; (void)out_size; (void)ws_size;
  const float* feat    = (const float*)d_in[0];
  const float* anchors = (const float*)d_in[1];
  const float* conv_w  = (const float*)d_in[2];
  const float* conv_b  = (const float*)d_in[3];
  const float* cls_w   = (const float*)d_in[4];
  const float* cls_b   = (const float*)d_in[5];
  const float* bbox_w  = (const float*)d_in[6];
  const float* bbox_b  = (const float*)d_in[7];
  float* out = (float*)d_out;

  char* ws = (char*)d_ws;
  unsigned short*     t_bf  = (unsigned short*)ws;                 // 4,194,304 B
  float*              breg  = (float*)(ws + 4194304);              // 3,932,160 B
  unsigned long long* keys  = (unsigned long long*)(ws + 8126464); // 2,097,152 B
  float*              iou   = (float*)(ws + 10223616);             // 8,000,000 B
  unsigned short*     wpack = (unsigned short*)(ws + 18223616);    // 1,179,648 B
  unsigned short*     hpack = (unsigned short*)(ws + 19403264);    //    73,728 B

  pack_conv_w<<<(72 * 16 * 512 + 255) / 256, 256, 0, stream>>>(conv_w, wpack);
  pack_head_w<<<(8 * 9 * 512 + 255) / 256, 256, 0, stream>>>(cls_w, bbox_w, hpack);
  zero_keys<<<(NIMG * SORTN + 255) / 256, 256, 0, stream>>>(keys);

  conv3x3_wmma<<<dim3(4, 128), 256, 0, stream>>>(feat, wpack, conv_b, t_bf);
  head_wmma<<<dim3(512, 9), 32, 0, stream>>>(t_bf, hpack, cls_b, bbox_b,
                                             breg, keys);
  for (int k = 2; k <= SORTN; k <<= 1)
    for (int j = k >> 1; j >= 1; j >>= 1)
      bitonic_kernel<<<(NIMG * SORTN) / 256, 256, 0, stream>>>(keys, k, j);
  decode_kernel<<<(NIMG * PRE_NMS + 255) / 256, 256, 0, stream>>>(keys, breg,
                                                                  anchors, out);
  long npairs = (long)NIMG * PRE_NMS * PRE_NMS;
  iou_kernel<<<(int)((npairs + 255) / 256), 256, 0, stream>>>(out, iou);
  nms_kernel<<<NIMG, 1024, 0, stream>>>(iou, out + NIMG * PRE_NMS * 6);
}
